// SpGAT_1803886265905
// MI455X (gfx1250) — matrix operand order repaired and checked by reading, not compiled, not run on hardware
//
#include <hip/hip_runtime.h>

// ---------------- types for WMMA (gfx1250, wave32) ----------------
typedef __attribute__((ext_vector_type(16))) __bf16 v16bf;
typedef __attribute__((ext_vector_type(8)))  __bf16 v8bf;
typedef __attribute__((ext_vector_type(8)))  float  v8f;

union AFrag { v16bf v16; v8bf v8[2]; };
union BFrag { v16bf v16; v8bf v8[2]; };

__device__ __forceinline__ v8f vzero8() {
  v8f z;
#pragma unroll
  for (int i = 0; i < 8; ++i) z[i] = 0.f;
  return z;
}

// ---------------- problem constants ----------------
#define NNODE   270
#define NFEAT   4096
#define NHID    1024
#define NHEADS  8
#define NCLASS  8
#define N_DRUG  175
#define N_MIC   95
#define MROW    320            // M padded to 5 * 64
#define KPAD2   288            // 9 * 32 (padded K for e@h)
#define HSTRIDE (MROW*NHID)    // per-head stride of h buffer
#define H2DIM   (NHEADS*NHID)
#define LDSB    (64*32)        // one B buffer: 64 cols x 32 k (bf16)

// ============================================================
// shared WMMA GEMM core: block tile 64M x 64N, 8 waves (each 32M x 16N),
// B (f32, row-major KxN slab) staged to LDS as bf16, transposed [n][k],
// 16B-chunk XOR swizzle, double buffered, one barrier per K step.
// ============================================================
__device__ __forceinline__ void gemm_tile_core(
    const __bf16* __restrict__ Abase, int strideA, int m0, int mg,
    const float* __restrict__ Bp, int n0, int K, bool doPrefetch,
    __bf16* Bt, int tid, v8f acc[2])
{
  const int lane   = tid & 31;
  const int row    = lane & 15;
  const int hiHalf = lane >> 4;          // 0 | 1
  const int akb    = hiHalf * 8;         // A: k {0..7,16..23} vs {8..15,24..31}
  const int ng     = (tid >> 5) & 3;     // wave's N group
  const int bn     = ng * 16 + row;      // B column (local 0..63)
  const int bsw    = (bn >> 2) & 3;      // read-side chunk swizzle
  const int c0     = hiHalf;             // chunks: lo half -> {0,2}, hi half -> {1,3}

  const int sn  = tid & 63;              // staging column
  const int sk0 = (tid >> 6) * 8;        // staging k base (0,8,16,24)
  const int ssw = (sn >> 2) & 3;         // write-side chunk swizzle

  const __bf16* ar0 = Abase + (size_t)(m0 + mg * 32 + row) * strideA;
  const __bf16* ar1 = ar0 + (size_t)16 * strideA;

  auto stage = [&](int buf, int kb) {
    __bf16* Bb = Bt + buf * LDSB;
#pragma unroll
    for (int p = 0; p < 4; ++p) {
      int k = sk0 + 2 * p;
      float v0 = Bp[(size_t)(kb + k)     * NHID + n0 + sn];
      float v1 = Bp[(size_t)(kb + k + 1) * NHID + n0 + sn];
      union { __bf16 h[2]; unsigned u; } pk;
      pk.h[0] = (__bf16)v0; pk.h[1] = (__bf16)v1;
      int idx = sn * 32 + (((k >> 3) ^ ssw) << 3) + (k & 7);
      *(unsigned*)&Bb[idx] = pk.u;
    }
  };

  stage(0, 0);
  __syncthreads();

  const int nIter = K >> 5;
  for (int it = 0; it < nIter; ++it) {
    const int cur = it & 1;
    const int kb  = it << 5;
    if (it + 1 < nIter) stage(cur ^ 1, kb + 32);
    if (doPrefetch && kb + 64 < K)
      __builtin_prefetch(&Bp[(size_t)(kb + 64 + sk0) * NHID + n0 + sn], 0, 0);

    const __bf16* Bb = Bt + cur * LDSB;
    BFrag b;
    b.v8[0] = *(const v8bf*)&Bb[bn * 32 + (((c0    ) ^ bsw) << 3)];
    b.v8[1] = *(const v8bf*)&Bb[bn * 32 + (((c0 + 2) ^ bsw) << 3)];

    AFrag a0, a1;
    a0.v8[0] = *(const v8bf*)(ar0 + kb + akb);
    a0.v8[1] = *(const v8bf*)(ar0 + kb + akb + 16);
    acc[0] = __builtin_amdgcn_wmma_f32_16x16x32_bf16(false, a0.v16, false, b.v16,
                                                     (short)0, acc[0], false, false);
    a1.v8[0] = *(const v8bf*)(ar1 + kb + akb);
    a1.v8[1] = *(const v8bf*)(ar1 + kb + akb + 16);
    acc[1] = __builtin_amdgcn_wmma_f32_16x16x32_bf16(false, a1.v16, false, b.v16,
                                                     (short)0, acc[1], false, false);
    __syncthreads();
  }
}

// ============================================================
// 1) convert x (270x4096 f32) -> xb (320x4096 bf16, pad rows = 0)
//    (zero pad rows make h rows 270..319 exactly 0 via the GEMM)
// ============================================================
__global__ void k_convert_x(const float* __restrict__ x, __bf16* __restrict__ xb) {
  int idx = blockIdx.x * 256 + threadIdx.x;
  if (idx >= MROW * NFEAT) return;
  int row = idx >> 12;   // / 4096
  xb[idx] = (row < NNODE) ? (__bf16)x[idx] : (__bf16)0.0f;
}

// ============================================================
// 2) h[hd] = xb @ W[hd]   grid (5, 16, 8), block 256
// ============================================================
__global__ void __launch_bounds__(256)
k_gemm_xw(const __bf16* __restrict__ xb, const float* __restrict__ W,
          float* __restrict__ h) {
  __shared__ __align__(32) __bf16 Bt[2 * LDSB];
  const int tid = threadIdx.x;
  const int m0 = blockIdx.x * 64, n0 = blockIdx.y * 64, hd = blockIdx.z;
  const int mg = (tid >> 5) >> 2;

  v8f acc[2] = {vzero8(), vzero8()};
  gemm_tile_core(xb, NFEAT, m0, mg, W + (size_t)hd * NFEAT * NHID, n0, NFEAT,
                 true, Bt, tid, acc);

  const int lane = tid & 31;
  const int row = lane & 15, hiHalf = lane >> 4, ng = (tid >> 5) & 3;
  const int ncol = n0 + ng * 16 + row;
  float* hp = h + (size_t)hd * HSTRIDE;
#pragma unroll
  for (int t = 0; t < 2; ++t) {
    const int mbase = m0 + mg * 32 + t * 16 + hiHalf * 8;
#pragma unroll
    for (int r = 0; r < 8; ++r)
      hp[(size_t)(mbase + r) * NHID + ncol] = acc[t][r];
  }
}

// ============================================================
// 3) per-row attention dots: src[i]=h_i.a_src, dst[i]=h_i.a_dst
// ============================================================
__global__ void __launch_bounds__(256)
k_dots(const float* __restrict__ h, const float* __restrict__ a_src,
       const float* __restrict__ a_dst, float* __restrict__ srcb,
       float* __restrict__ dstb) {
  __shared__ float ss[256], sd[256];
  const int i = blockIdx.x, hd = blockIdx.y, tid = threadIdx.x;
  const float* hr = h + (size_t)hd * HSTRIDE + (size_t)i * NHID;
  const float* as = a_src + hd * NHID;
  const float* ad = a_dst + hd * NHID;
  float s = 0.f, d = 0.f;
  for (int k = tid; k < NHID; k += 256) { float v = hr[k]; s += v * as[k]; d += v * ad[k]; }
  ss[tid] = s; sd[tid] = d; __syncthreads();
  for (int st = 128; st > 0; st >>= 1) {
    if (tid < st) { ss[tid] += ss[tid + st]; sd[tid] += sd[tid + st]; }
    __syncthreads();
  }
  if (tid == 0) { srcb[hd * MROW + i] = ss[0]; dstb[hd * MROW + i] = sd[0]; }
}

// ============================================================
// 4) e[hd][i][j] = exp(-lrelu(src_i+dst_j)) * mask, bf16, padded 320x288
//    plus rowsum.  grid (320, 8), block 128
// ============================================================
__global__ void __launch_bounds__(128)
k_edge(const float* __restrict__ srcb, const float* __restrict__ dstb,
       const int* __restrict__ adj, __bf16* __restrict__ eb,
       float* __restrict__ rowsum) {
  __shared__ float sm[128];
  const int i = blockIdx.x, hd = blockIdx.y, tid = threadIdx.x;
  const float si = (i < NNODE) ? srcb[hd * MROW + i] : 0.f;
  float local = 0.f;
  for (int j = tid; j < KPAD2; j += 128) {
    float ev = 0.f;
    if (i < NNODE && j < NNODE) {
      float l  = si + dstb[hd * MROW + j];
      float lr = (l > 0.f) ? l : 0.2f * l;
      ev = expf(-lr) * (float)adj[i * NNODE + j];
    }
    eb[(size_t)hd * (MROW * KPAD2) + (size_t)i * KPAD2 + j] = (__bf16)ev;
    local += ev;
  }
  sm[tid] = local; __syncthreads();
  for (int st = 64; st > 0; st >>= 1) {
    if (tid < st) sm[tid] += sm[tid + st];
    __syncthreads();
  }
  if (tid == 0) rowsum[hd * MROW + i] = sm[0];
}

// ============================================================
// 5) h2[i][hd*1024+n] = elu( (e @ h)[i][n] / rowsum[i] )  grid (5,16,8)
// ============================================================
__global__ void __launch_bounds__(256)
k_gemm_eh(const __bf16* __restrict__ eb, const float* __restrict__ h,
          const float* __restrict__ rowsum, float* __restrict__ h2) {
  __shared__ __align__(32) __bf16 Bt[2 * LDSB];
  const int tid = threadIdx.x;
  const int m0 = blockIdx.x * 64, n0 = blockIdx.y * 64, hd = blockIdx.z;
  const int mg = (tid >> 5) >> 2;

  v8f acc[2] = {vzero8(), vzero8()};
  gemm_tile_core(eb + (size_t)hd * (MROW * KPAD2), KPAD2, m0, mg,
                 h + (size_t)hd * HSTRIDE, n0, KPAD2, false, Bt, tid, acc);

  const int lane = tid & 31;
  const int row = lane & 15, hiHalf = lane >> 4, ng = (tid >> 5) & 3;
  const int ncol = n0 + ng * 16 + row;
#pragma unroll
  for (int t = 0; t < 2; ++t) {
    const int mbase = m0 + mg * 32 + t * 16 + hiHalf * 8;
#pragma unroll
    for (int r = 0; r < 8; ++r) {
      int i = mbase + r;
      if (i < NNODE) {
        float v = acc[t][r] / rowsum[hd * MROW + i];
        v = (v > 0.f) ? v : expm1f(v);               // ELU
        h2[(size_t)i * H2DIM + hd * NHID + ncol] = v;
      }
    }
  }
}

// ============================================================
// 6) output layer linear (N=8): out_h = h2 @ W_out, plus s2/d2 dots
// ============================================================
__global__ void __launch_bounds__(256)
k_outlayer(const float* __restrict__ h2, const float* __restrict__ W_out,
           const float* __restrict__ a_osrc, const float* __restrict__ a_odst,
           float* __restrict__ out_h, float* __restrict__ s2, float* __restrict__ d2) {
  __shared__ float red[NCLASS * 256];
  const int i = blockIdx.x, tid = threadIdx.x;
  float acc[NCLASS];
#pragma unroll
  for (int c = 0; c < NCLASS; ++c) acc[c] = 0.f;
  const float* hr = h2 + (size_t)i * H2DIM;
  for (int k = tid; k < H2DIM; k += 256) {
    float v = hr[k];
    const float* wr = W_out + (size_t)k * NCLASS;
#pragma unroll
    for (int c = 0; c < NCLASS; ++c) acc[c] += v * wr[c];
  }
#pragma unroll
  for (int c = 0; c < NCLASS; ++c) red[c * 256 + tid] = acc[c];
  __syncthreads();
  for (int st = 128; st > 0; st >>= 1) {
    if (tid < st)
#pragma unroll
      for (int c = 0; c < NCLASS; ++c) red[c * 256 + tid] += red[c * 256 + tid + st];
    __syncthreads();
  }
  if (tid == 0) {
    float sv = 0.f, dv = 0.f;
#pragma unroll
    for (int c = 0; c < NCLASS; ++c) {
      float o = red[c * 256];
      out_h[i * NCLASS + c] = o;
      sv += o * a_osrc[c];
      dv += o * a_odst[c];
    }
    s2[i] = sv; d2[i] = dv;
  }
}

// ============================================================
// 7) second attention layer + ELU.  grid 270, block 128
// ============================================================
__global__ void __launch_bounds__(128)
k_attn2(const int* __restrict__ adj, const float* __restrict__ s2,
        const float* __restrict__ d2, const float* __restrict__ out_h,
        float* __restrict__ out2) {
  __shared__ float red[(NCLASS + 1) * 128];
  const int i = blockIdx.x, tid = threadIdx.x;
  float num[NCLASS];
#pragma unroll
  for (int c = 0; c < NCLASS; ++c) num[c] = 0.f;
  float den = 0.f;
  const float si = s2[i];
  for (int j = tid; j < NNODE; j += 128) {
    float l  = si + d2[j];
    float lr = (l > 0.f) ? l : 0.2f * l;
    float ev = expf(-lr) * (float)adj[i * NNODE + j];
    den += ev;
    const float* oj = out_h + j * NCLASS;
#pragma unroll
    for (int c = 0; c < NCLASS; ++c) num[c] += ev * oj[c];
  }
#pragma unroll
  for (int c = 0; c < NCLASS; ++c) red[c * 128 + tid] = num[c];
  red[NCLASS * 128 + tid] = den;
  __syncthreads();
  for (int st = 64; st > 0; st >>= 1) {
    if (tid < st)
#pragma unroll
      for (int c = 0; c <= NCLASS; ++c) red[c * 128 + tid] += red[c * 128 + tid + st];
    __syncthreads();
  }
  if (tid == 0) {
    float dn = red[NCLASS * 128];
#pragma unroll
    for (int c = 0; c < NCLASS; ++c) {
      float v = red[c * 128] / dn;
      v = (v > 0.f) ? v : expm1f(v);
      out2[i * NCLASS + c] = v;
    }
  }
}

// ============================================================
// 8) final: res[i][j] = drug[i] @ alpha1 @ mic[j]
// ============================================================
__global__ void __launch_bounds__(256)
k_final(const float* __restrict__ out2, const float* __restrict__ alpha1,
        float* __restrict__ res) {
  __shared__ float al[NCLASS * NCLASS];
  if (threadIdx.x < NCLASS * NCLASS) al[threadIdx.x] = alpha1[threadIdx.x];
  __syncthreads();
  int idx = blockIdx.x * 256 + threadIdx.x;
  if (idx >= N_DRUG * N_MIC) return;
  int i = idx / N_MIC, j = idx % N_MIC;
  const float* dr = out2 + i * NCLASS;
  const float* mi = out2 + (N_DRUG + j) * NCLASS;
  float s = 0.f;
#pragma unroll
  for (int d = 0; d < NCLASS; ++d) {
    float t = 0.f;
#pragma unroll
    for (int c = 0; c < NCLASS; ++c) t += dr[c] * al[c * NCLASS + d];
    s += t * mi[d];
  }
  res[idx] = s;
}

// ============================================================
extern "C" void kernel_launch(void* const* d_in, const int* in_sizes, int n_in,
                              void* d_out, int out_size, void* d_ws, size_t ws_size,
                              hipStream_t stream) {
  const float* x      = (const float*)d_in[0];
  const int*   adj    = (const int*)  d_in[1];
  const float* W      = (const float*)d_in[2];
  const float* a_src  = (const float*)d_in[3];
  const float* a_dst  = (const float*)d_in[4];
  const float* W_out  = (const float*)d_in[5];
  const float* a_osrc = (const float*)d_in[6];
  const float* a_odst = (const float*)d_in[7];
  const float* alpha1 = (const float*)d_in[8];
  float* out = (float*)d_out;

  char* ws = (char*)d_ws;
  size_t off = 0;
  auto alloc = [&](size_t bytes) -> char* {
    char* p = ws + off;
    off = (off + bytes + 255) & ~(size_t)255;
    return p;
  };
  __bf16* xb    = (__bf16*)alloc((size_t)MROW * NFEAT * 2);
  float*  h     = (float*) alloc((size_t)NHEADS * HSTRIDE * 4);
  float*  srcb  = (float*) alloc((size_t)NHEADS * MROW * 4);
  float*  dstb  = (float*) alloc((size_t)NHEADS * MROW * 4);
  __bf16* eb    = (__bf16*)alloc((size_t)NHEADS * MROW * KPAD2 * 2);
  float*  rsum  = (float*) alloc((size_t)NHEADS * MROW * 4);
  float*  h2    = (float*) alloc((size_t)NNODE * H2DIM * 4);
  float*  out_h = (float*) alloc((size_t)NNODE * NCLASS * 4);
  float*  s2    = (float*) alloc((size_t)NNODE * 4);
  float*  d2    = (float*) alloc((size_t)NNODE * 4);
  float*  out2  = (float*) alloc((size_t)NNODE * NCLASS * 4);

  k_convert_x<<<(MROW * NFEAT + 255) / 256, 256, 0, stream>>>(x, xb);
  k_gemm_xw<<<dim3(5, 16, 8), 256, 0, stream>>>(xb, W, h);
  k_dots<<<dim3(NNODE, NHEADS), 256, 0, stream>>>(h, a_src, a_dst, srcb, dstb);
  k_edge<<<dim3(MROW, NHEADS), 128, 0, stream>>>(srcb, dstb, adj, eb, rsum);
  k_gemm_eh<<<dim3(5, 16, 8), 256, 0, stream>>>(eb, h, rsum, h2);
  k_outlayer<<<NNODE, 256, 0, stream>>>(h2, W_out, a_osrc, a_odst, out_h, s2, d2);
  k_attn2<<<NNODE, 128, 0, stream>>>(adj, s2, d2, out_h, out2);
  k_final<<<(N_DRUG * N_MIC + 255) / 256, 256, 0, stream>>>(out2, alpha1, out);
}
